// MultiscreenModel_89309549953336
// MI455X (gfx1250) — compile-verified, matrix operand order checked by reading, not run on hardware
//
#include <hip/hip_runtime.h>
#include <hip/hip_bf16.h>
#include <math.h>

// ---------------- model constants ----------------
#define DE_    1024
#define DK_    64
#define DV_    64
#define NH_    16
#define NL_    2
#define TSEQ   2048
#define VOCAB_ 32000
#define WTH_F  4096.0f
#define PI_F   3.14159265358979f

typedef __bf16 bf16;
typedef __attribute__((ext_vector_type(8)))  __bf16 v8bf;
typedef __attribute__((ext_vector_type(16))) __bf16 v16bf;
typedef __attribute__((ext_vector_type(8)))  float  v8f;

typedef __attribute__((ext_vector_type(4))) unsigned int u32x4;
typedef __attribute__((ext_vector_type(8))) int          i32x8;
typedef __attribute__((ext_vector_type(4))) int          i32x4;

#if defined(__has_builtin)
#if __has_builtin(__builtin_amdgcn_tensor_load_to_lds) && __has_builtin(__builtin_amdgcn_s_wait_tensorcnt)
#define HAVE_TDM 1
#endif
#endif
#ifndef HAVE_TDM
#define HAVE_TDM 0
#endif

static __device__ __forceinline__ v16bf make_frag(v8bf lo, v8bf hi) {
  return __builtin_shufflevector(lo, hi, 0,1,2,3,4,5,6,7,8,9,10,11,12,13,14,15);
}
static __device__ __forceinline__ v8f wmma_bf16(v16bf a, v16bf b, v8f c) {
  // D = A(16x32) * B(32x16) + C, f32 accumulate
  return __builtin_amdgcn_wmma_f32_16x16x32_bf16(false, a, false, b, (short)0, c, false, false);
}
static __device__ __forceinline__ v8f v8f_zero() {
  v8f z = {0.f,0.f,0.f,0.f,0.f,0.f,0.f,0.f};
  return z;
}

#if HAVE_TDM
// LDS byte offset of a __shared__ object: generic LDS addresses carry the
// offset in the low 32 bits (aperture lives in addr[63:32], ISA 10.2).
static __device__ __forceinline__ unsigned lds_off(const void* p) {
  return (unsigned)(unsigned long long)p;
}
// Issue a 2-D TDM tile load Global->LDS (D# per ISA 08_async_tensor 8.3/8.4).
// data_size = 2 bytes (bf16). Optional LDS padding: pad_interval code 3 = every
// 64 bytes stored, pad_amount code 3 = 4 DWORDs -> LDS row stride 80 B (LDA=40).
static __device__ __forceinline__ void tdm_load_2d(
    const void* gptr, unsigned lds_byte,
    unsigned tile_k, unsigned tile_rows,
    unsigned tensor_k, unsigned tensor_rows,
    unsigned row_stride_elems, bool pad_en)
{
  const unsigned long long ga = (unsigned long long)gptr;
  u32x4 g0;
  g0[0] = 1u;                                            // count=1, user descriptor
  g0[1] = lds_byte;                                      // lds_addr
  g0[2] = (unsigned)(ga & 0xFFFFFFFFu);                  // global_addr[31:0]
  g0[3] = (unsigned)((ga >> 32) & 0x1FFFFFFu) | (2u << 30); // addr[56:32] | type=2
  i32x8 g1;
  unsigned w0 = (1u << 16);                              // data_size=1 (2 bytes)
  if (pad_en) w0 |= (1u << 20) | (3u << 22) | (3u << 25); // pad_en, interval=64B, amount=4dw
  g1[0] = (int)w0;
  g1[1] = (int)((tensor_k & 0xFFFFu) << 16);             // tensor_dim0[15:0]
  g1[2] = (int)(((tensor_k >> 16) & 0xFFFFu) | ((tensor_rows & 0xFFFFu) << 16));
  g1[3] = (int)(((tensor_rows >> 16) & 0xFFFFu) | ((tile_k & 0xFFFFu) << 16)); // tile_dim0
  g1[4] = (int)(tile_rows & 0xFFFFu);                    // tile_dim1 (tile_dim2=0)
  g1[5] = (int)row_stride_elems;                         // tensor_dim0_stride[31:0]
  g1[6] = 0;
  g1[7] = 0;
  i32x4 z4 = {0, 0, 0, 0};
#if __clang_major__ >= 23
  i32x8 z8 = {0, 0, 0, 0, 0, 0, 0, 0};
  __builtin_amdgcn_tensor_load_to_lds(g0, g1, z4, z4, z8, 0);
#else
  __builtin_amdgcn_tensor_load_to_lds(g0, g1, z4, z4, 0);
#endif
}
#endif // HAVE_TDM

// ---------------- embed row-normalize -> bf16 ----------------
__global__ __launch_bounds__(256) void k_norm_rows(
    const float* __restrict__ W, bf16* __restrict__ Wn, int cols)
{
  __shared__ float red[8];
  const int row = blockIdx.x;
  const float* src = W + (size_t)row * cols;
  float ss = 0.f;
  for (int c = threadIdx.x; c < cols; c += blockDim.x) { float v = src[c]; ss += v * v; }
  for (int off = 16; off > 0; off >>= 1) ss += __shfl_down(ss, off, 32);
  if ((threadIdx.x & 31) == 0) red[threadIdx.x >> 5] = ss;
  __syncthreads();
  if (threadIdx.x == 0) {
    float t = 0.f;
    for (int i = 0; i < 8; ++i) t += red[i];
    red[0] = 1.0f / fmaxf(sqrtf(t), 1e-12f);
  }
  __syncthreads();
  const float inv = red[0];
  bf16* dst = Wn + (size_t)row * cols;
  for (int c = threadIdx.x; c < cols; c += blockDim.x) dst[c] = (bf16)(src[c] * inv);
}

// ---------------- f32 -> bf16 bulk convert ----------------
__global__ __launch_bounds__(256) void k_f32_to_bf16(
    const float* __restrict__ in, bf16* __restrict__ out, size_t n)
{
  size_t i = (size_t)blockIdx.x * blockDim.x + threadIdx.x;
  if (i < n) out[i] = (bf16)in[i];
}

// ---------------- embedding gather ----------------
__global__ __launch_bounds__(256) void k_gather_embed(
    const int* __restrict__ ids, const bf16* __restrict__ Wn,
    const float* __restrict__ sE,
    float* __restrict__ x, bf16* __restrict__ xb)
{
  const int t = blockIdx.x;
  const int id = ids[t];
  const float s = expf(*sE);
  const bf16* src = Wn + (size_t)id * DE_;
  for (int c = threadIdx.x; c < DE_; c += blockDim.x) {
    float v = (float)src[c] * s;
    x[(size_t)t * DE_ + c]  = v;
    xb[(size_t)t * DE_ + c] = (bf16)v;
  }
}

// ---------------- generic bf16 WMMA GEMM: C[M,N] = scale * A[M,K] * B[N,K]^T (+resid) ----------------
__global__ __launch_bounds__(256) void k_gemm_bf16(
    const bf16* __restrict__ A,        // M x K row-major
    const bf16* __restrict__ B,        // N x K row-major (weight layout)
    float* __restrict__ C,             // M x N
    bf16* __restrict__ Cbf,            // optional bf16 mirror (nullable)
    const float* __restrict__ resid,   // optional residual add, M x N (nullable)
    const float* __restrict__ scale_log, // optional: scale = exp(*scale_log)
    int M, int N, int K)
{
  constexpr int TM = 128, TN = 64, TK = 32, LDA = 40; // 80B LDS rows: 16B-aligned
#if HAVE_TDM
  __shared__ bf16 As[2][TM * LDA];
  __shared__ bf16 Bs[2][TN * LDA];
#else
  __shared__ bf16 As[1][TM * LDA];
  __shared__ bf16 Bs[1][TN * LDA];
#endif

  const int tid  = threadIdx.x;
  const int lane = tid & 31;
  const int wave = tid >> 5;           // 0..7
  const int m0   = blockIdx.x * TM;
  const int n0   = blockIdx.y * TN;

  const float scale = scale_log ? expf(*scale_log) : 1.0f;

  const int rA    = lane & 15;
  const int khalf = (lane >> 4) * 8;   // bf16 A/B fragment K-half select (ISA 7.12.2)

  v8f acc0 = v8f_zero(), acc1 = v8f_zero(), acc2 = v8f_zero(), acc3 = v8f_zero();

  auto compute_tile = [&](const bf16* Asb, const bf16* Bsb) {
    const int arow = wave * 16 + rA;
    v16bf af = make_frag(*(const v8bf*)(Asb + arow * LDA + khalf),
                         *(const v8bf*)(Asb + arow * LDA + 16 + khalf));
    v16bf b0 = make_frag(*(const v8bf*)(Bsb + ( 0 + rA) * LDA + khalf),
                         *(const v8bf*)(Bsb + ( 0 + rA) * LDA + 16 + khalf));
    v16bf b1 = make_frag(*(const v8bf*)(Bsb + (16 + rA) * LDA + khalf),
                         *(const v8bf*)(Bsb + (16 + rA) * LDA + 16 + khalf));
    v16bf b2 = make_frag(*(const v8bf*)(Bsb + (32 + rA) * LDA + khalf),
                         *(const v8bf*)(Bsb + (32 + rA) * LDA + 16 + khalf));
    v16bf b3 = make_frag(*(const v8bf*)(Bsb + (48 + rA) * LDA + khalf),
                         *(const v8bf*)(Bsb + (48 + rA) * LDA + 16 + khalf));
    acc0 = wmma_bf16(af, b0, acc0);
    acc1 = wmma_bf16(af, b1, acc1);
    acc2 = wmma_bf16(af, b2, acc2);
    acc3 = wmma_bf16(af, b3, acc3);
  };

#if HAVE_TDM
  // -------- TDM double-buffered pipeline: wave 0 drives the Tensor Data Mover --------
  const int nk = K / TK;
  if (wave == 0) {
    tdm_load_2d(A + (size_t)m0 * K, lds_off(&As[0][0]), TK, TM, (unsigned)K, TM, (unsigned)K, true);
    tdm_load_2d(B + (size_t)n0 * K, lds_off(&Bs[0][0]), TK, TN, (unsigned)K, TN, (unsigned)K, true);
  }
  for (int kk = 0; kk < nk; ++kk) {
    const int buf = kk & 1;
    if (wave == 0) {
      if (kk + 1 < nk) {
        const int k1 = (kk + 1) * TK;
        tdm_load_2d(A + (size_t)m0 * K + k1, lds_off(&As[buf ^ 1][0]), TK, TM, (unsigned)K, TM, (unsigned)K, true);
        tdm_load_2d(B + (size_t)n0 * K + k1, lds_off(&Bs[buf ^ 1][0]), TK, TN, (unsigned)K, TN, (unsigned)K, true);
        __builtin_amdgcn_s_wait_tensorcnt(2);  // stage kk's two descriptors done (in-order)
      } else {
        __builtin_amdgcn_s_wait_tensorcnt(0);
      }
    }
    __syncthreads();
    compute_tile(&As[buf][0], &Bs[buf][0]);
    __syncthreads();
  }
#else
  // -------- fallback: cooperative vector loads through LDS --------
  const int chunk = tid & 3;           // 4 x 8 bf16 = 32 K per row
  for (int k0 = 0; k0 < K; k0 += TK) {
    for (int r = tid >> 2; r < TM; r += 64) {
      v8bf v = *(const v8bf*)(A + (size_t)(m0 + r) * K + k0 + chunk * 8);
      *(v8bf*)(&As[0][0] + r * LDA + chunk * 8) = v;
    }
    {
      int rB = tid >> 2; // 0..63
      v8bf v = *(const v8bf*)(B + (size_t)(n0 + rB) * K + k0 + chunk * 8);
      *(v8bf*)(&Bs[0][0] + rB * LDA + chunk * 8) = v;
    }
    __syncthreads();
    if (k0 + TK < K) { // pull next K-tile toward L2 (global_prefetch_b8)
      __builtin_prefetch(A + (size_t)(m0 + (tid >> 1)) * K + k0 + TK, 0, 3);
      __builtin_prefetch(B + (size_t)(n0 + (tid >> 2)) * K + k0 + TK, 0, 3);
    }
    compute_tile(&As[0][0], &Bs[0][0]);
    __syncthreads();
  }
#endif

  // epilogue: lane holds col = lane&15, rows e + 8*(lane>=16) (ISA 7.12.2 C/D layout)
  const int mrow0 = m0 + wave * 16 + ((lane >> 4) << 3);
  const int col   = lane & 15;
  v8f accs[4] = {acc0, acc1, acc2, acc3};
  #pragma unroll
  for (int ns = 0; ns < 4; ++ns) {
    const int n = n0 + ns * 16 + col;
    #pragma unroll
    for (int e = 0; e < 8; ++e) {
      const size_t idx = (size_t)(mrow0 + e) * N + n;
      float v = accs[ns][e] * scale;
      if (resid) v += resid[idx];
      C[idx] = v;
      if (Cbf) Cbf[idx] = (bf16)v;
    }
  }
}

// ---------------- q/k/v: unit-normalize + pair rotation -> bf16 ----------------
__global__ __launch_bounds__(64) void k_qkv_finish(
    const float* __restrict__ qf, const float* __restrict__ kf, const float* __restrict__ vf,
    const float* __restrict__ sw,
    bf16* __restrict__ qb, bf16* __restrict__ kb, bf16* __restrict__ vb)
{
  const int t = blockIdx.x, h = blockIdx.y, d = threadIdx.x;
  const size_t base = ((size_t)t * NH_ + h) * 64 + d;
  __shared__ float sh[3][64];
  const float q = qf[base], k = kf[base], v = vf[base];
  sh[0][d] = q * q; sh[1][d] = k * k; sh[2][d] = v * v;
  __syncthreads();
  for (int off = 32; off > 0; off >>= 1) {
    if (d < off) { sh[0][d] += sh[0][d+off]; sh[1][d] += sh[1][d+off]; sh[2][d] += sh[2][d+off]; }
    __syncthreads();
  }
  const float iq = 1.0f / fmaxf(sqrtf(sh[0][0]), 1e-12f);
  const float ik = 1.0f / fmaxf(sqrtf(sh[1][0]), 1e-12f);
  const float iv = 1.0f / fmaxf(sqrtf(sh[2][0]), 1e-12f);
  __syncthreads();
  const float qn = q * iq, kn = k * ik;
  sh[0][d] = qn; sh[1][d] = kn;
  __syncthreads();
  const float w   = expf(sw[h]) + 1.0f;
  const float phi = (w < WTH_F) ? 0.5f * (cosf(PI_F * w / WTH_F) + 1.0f) : 0.0f;
  const float ang = (float)t * (phi / w);
  const float ca = cosf(ang), sa = sinf(ang);
  float qo = qn, ko = kn;
  if (d == 0)      { qo = sh[0][0]*ca - sh[0][1]*sa; ko = sh[1][0]*ca - sh[1][1]*sa; }
  else if (d == 1) { qo = sh[0][0]*sa + sh[0][1]*ca; ko = sh[1][0]*sa + sh[1][1]*ca; }
  qb[base] = (bf16)qo; kb[base] = (bf16)ko; vb[base] = (bf16)(v * iv);
}

// ---------------- windowed attention: one wave per (16-row tile, head) ----------------
__global__ __launch_bounds__(32) void k_attn(
    const bf16* __restrict__ qb, const bf16* __restrict__ kb, const bf16* __restrict__ vb,
    const float* __restrict__ sw, const float* __restrict__ sr,
    float* __restrict__ hf)
{
  __shared__ bf16 rhoS[16 * 40];   // 16 rows x 32 cols, 80B row stride (16B aligned)
  __shared__ bf16 vS[32 * 80];     // 32 rows x 64 cols, 160B row stride

  const int lane = threadIdx.x;
  const int it   = blockIdx.x;     // 16-row tile of queries
  const int h    = blockIdx.y;
  const int i0   = it * 16;
  const float w    = expf(sw[h]) + 1.0f;
  const float r    = expf(sr[h]) + 1.0f;
  const float invw = 1.0f / w;

  const int rA    = lane & 15;
  const int khalf = (lane >> 4) * 8;
  const int m0l   = (lane >> 4) << 3;

  // q row fragments (K = 64 -> two 16x32 A fragments)
  const bf16* qrow = qb + ((size_t)(i0 + rA) * NH_ + h) * 64;
  const v16bf aq0 = make_frag(*(const v8bf*)(qrow + khalf),      *(const v8bf*)(qrow + 16 + khalf));
  const v16bf aq1 = make_frag(*(const v8bf*)(qrow + 32 + khalf), *(const v8bf*)(qrow + 48 + khalf));

  v8f hacc[4] = {v8f_zero(), v8f_zero(), v8f_zero(), v8f_zero()};

  const int jtMax = (i0 + 15) >> 5;                    // causal upper bound
  int jlo = i0 - (int)w; if (jlo < 0) jlo = 0;         // window lower bound (mask handles exact)
  const int jtMin = jlo >> 5;

  for (int jt = jtMin; jt <= jtMax; ++jt) {
    const int j0 = jt * 32;

    // stage v tile (32 x 64) into LDS, one row per lane
    {
      const bf16* vrow = vb + ((size_t)(j0 + lane) * NH_ + h) * 64;
      #pragma unroll
      for (int c = 0; c < 64; c += 8)
        *(v8bf*)(vS + lane * 80 + c) = *(const v8bf*)(vrow + c);
    }

    #pragma unroll
    for (int jsub = 0; jsub < 2; ++jsub) {
      const int jc0 = j0 + jsub * 16;
      const bf16* krow = kb + ((size_t)(jc0 + rA) * NH_ + h) * 64;
      const v16bf bk0 = make_frag(*(const v8bf*)(krow + khalf),      *(const v8bf*)(krow + 16 + khalf));
      const v16bf bk1 = make_frag(*(const v8bf*)(krow + 32 + khalf), *(const v8bf*)(krow + 48 + khalf));
      v8f s = v8f_zero();
      s = wmma_bf16(aq0, bk0, s);
      s = wmma_bf16(aq1, bk1, s);

      const int j = jc0 + rA;
      #pragma unroll
      for (int e = 0; e < 8; ++e) {
        const int i   = i0 + m0l + e;
        const int rel = j - i;
        float rho = 0.0f;
        if (rel <= 0 && (float)(-rel) < w) {
          const float mask = 0.5f * (cosf(PI_F * (float)rel * invw) + 1.0f);
          const float tt = fmaxf(1.0f - r * (1.0f - s[e]), 0.0f);
          rho = tt * tt * mask;
        }
        rhoS[(m0l + e) * 40 + jsub * 16 + rA] = (bf16)rho;
      }
    }
    __syncthreads();

    // repack rho (16x32) into A-fragment layout, v columns from LDS as B fragments
    const v16bf ar = make_frag(*(const v8bf*)(rhoS + rA * 40 + khalf),
                               *(const v8bf*)(rhoS + rA * 40 + 16 + khalf));
    #pragma unroll
    for (int ns = 0; ns < 4; ++ns) {
      const int n = ns * 16 + rA;
      v16bf bv;
      #pragma unroll
      for (int e = 0; e < 8; ++e) {
        bv[e]     = vS[(khalf + e) * 80 + n];
        bv[e + 8] = vS[(16 + khalf + e) * 80 + n];
      }
      hacc[ns] = wmma_bf16(ar, bv, hacc[ns]);
    }
    __syncthreads();
  }

  #pragma unroll
  for (int ns = 0; ns < 4; ++ns)
    #pragma unroll
    for (int e = 0; e < 8; ++e)
      hf[((size_t)(i0 + m0l + e) * NH_ + h) * 64 + ns * 16 + rA] = hacc[ns][e];
}

// ---------------- gate: u = tanh(|h|)/|h| * h ; hgo = u * tanh(silu(g)) * exp(sO) ----------------
__global__ __launch_bounds__(64) void k_gate(
    const float* __restrict__ hf, const float* __restrict__ gf,
    const float* __restrict__ sO, bf16* __restrict__ hgo)
{
  const int t = blockIdx.x, h = blockIdx.y, d = threadIdx.x;
  const size_t base = ((size_t)t * NH_ + h) * 64 + d;
  __shared__ float sh[64];
  const float hv = hf[base];
  sh[d] = hv * hv;
  __syncthreads();
  for (int off = 32; off > 0; off >>= 1) {
    if (d < off) sh[d] += sh[d + off];
    __syncthreads();
  }
  const float hn = fmaxf(sqrtf(sh[0]), 1e-8f);
  const float u  = tanhf(hn) / hn * hv;
  const float g  = gf[base];
  const float ghat = tanhf(g / (1.0f + expf(-g)));
  hgo[base] = (bf16)(u * ghat * expf(sO[h]));
}

// ---------------- orchestration ----------------
extern "C" void kernel_launch(void* const* d_in, const int* in_sizes, int n_in,
                              void* d_out, int out_size, void* d_ws, size_t ws_size,
                              hipStream_t stream)
{
  (void)in_sizes; (void)n_in; (void)out_size; (void)ws_size;
  const int*   ids   = (const int*)d_in[0];
  const float* embed = (const float*)d_in[1];
  const float* sE    = (const float*)d_in[2];
  const float* sF    = (const float*)d_in[3];
  const float* Wq    = (const float*)d_in[4];
  const float* Wk    = (const float*)d_in[5];
  const float* Wv    = (const float*)d_in[6];
  const float* Wg    = (const float*)d_in[7];
  const float* Wo    = (const float*)d_in[8];
  const float* sw    = (const float*)d_in[9];
  const float* sr    = (const float*)d_in[10];
  const float* sO    = (const float*)d_in[11];

  char* ws = (char*)d_ws;
  auto alloc = [&](size_t bytes) -> char* {
    char* p = ws; ws += (bytes + 255) & ~(size_t)255; return p;
  };
  const size_t WSZ = (size_t)NL_ * DE_ * DE_;   // per-tensor weight elements

  bf16*  wn   = (bf16*)alloc((size_t)VOCAB_ * DE_ * sizeof(bf16));
  bf16*  wqb  = (bf16*)alloc(WSZ * sizeof(bf16));
  bf16*  wkb  = (bf16*)alloc(WSZ * sizeof(bf16));
  bf16*  wvb  = (bf16*)alloc(WSZ * sizeof(bf16));
  bf16*  wgb  = (bf16*)alloc(WSZ * sizeof(bf16));
  bf16*  wob  = (bf16*)alloc(WSZ * sizeof(bf16));
  float* x    = (float*)alloc((size_t)TSEQ * DE_ * sizeof(float));
  bf16*  xb   = (bf16*)alloc((size_t)TSEQ * DE_ * sizeof(bf16));
  float* qf   = (float*)alloc((size_t)TSEQ * DE_ * sizeof(float));
  float* kf   = (float*)alloc((size_t)TSEQ * DE_ * sizeof(float));
  float* vf   = (float*)alloc((size_t)TSEQ * DE_ * sizeof(float));
  float* gf   = (float*)alloc((size_t)TSEQ * DE_ * sizeof(float));
  bf16*  qbb  = (bf16*)alloc((size_t)TSEQ * DE_ * sizeof(bf16));
  bf16*  kbb  = (bf16*)alloc((size_t)TSEQ * DE_ * sizeof(bf16));
  bf16*  vbb  = (bf16*)alloc((size_t)TSEQ * DE_ * sizeof(bf16));
  float* hf   = (float*)alloc((size_t)TSEQ * DE_ * sizeof(float));
  bf16*  hgob = (bf16*)alloc((size_t)TSEQ * DE_ * sizeof(bf16));

  // 1) normalized embedding table in bf16 (shared by gather and logits GEMM)
  k_norm_rows<<<VOCAB_, 256, 0, stream>>>(embed, wn, DE_);

  // 2) weights -> bf16 (one pass; ~21 MB total, L2-resident afterwards)
  const int cgrid = (int)((WSZ + 255) / 256);
  k_f32_to_bf16<<<cgrid, 256, 0, stream>>>(Wq, wqb, WSZ);
  k_f32_to_bf16<<<cgrid, 256, 0, stream>>>(Wk, wkb, WSZ);
  k_f32_to_bf16<<<cgrid, 256, 0, stream>>>(Wv, wvb, WSZ);
  k_f32_to_bf16<<<cgrid, 256, 0, stream>>>(Wg, wgb, WSZ);
  k_f32_to_bf16<<<cgrid, 256, 0, stream>>>(Wo, wob, WSZ);

  // 3) x = unit(embed)[ids] * exp(s_E)
  k_gather_embed<<<TSEQ, 256, 0, stream>>>(ids, wn, sE, x, xb);

  const dim3 gProj(TSEQ / 128, DE_ / 64);     // (16,16)
  for (int l = 0; l < NL_; ++l) {
    const size_t wofs = (size_t)l * DE_ * DE_;
    const float* sw_l = sw + l * NH_;
    const float* sr_l = sr + l * NH_;
    const float* sO_l = sO + l * NH_;

    // projections: q/k/v/g (f32 out for the normalize/rotate epilogues)
    k_gemm_bf16<<<gProj, 256, 0, stream>>>(xb, wqb + wofs, qf, nullptr, nullptr, nullptr, TSEQ, DE_, DE_);
    k_gemm_bf16<<<gProj, 256, 0, stream>>>(xb, wkb + wofs, kf, nullptr, nullptr, nullptr, TSEQ, DE_, DE_);
    k_gemm_bf16<<<gProj, 256, 0, stream>>>(xb, wvb + wofs, vf, nullptr, nullptr, nullptr, TSEQ, DE_, DE_);
    k_gemm_bf16<<<gProj, 256, 0, stream>>>(xb, wgb + wofs, gf, nullptr, nullptr, nullptr, TSEQ, DE_, DE_);

    // unit-normalize + rotation -> bf16
    k_qkv_finish<<<dim3(TSEQ, NH_), 64, 0, stream>>>(qf, kf, vf, sw_l, qbb, kbb, vbb);

    // windowed attention, WMMA sim + WMMA rho*v
    k_attn<<<dim3(TSEQ / 16, NH_), 32, 0, stream>>>(qbb, kbb, vbb, sw_l, sr_l, hf);

    // norm-clamp + gating -> bf16
    k_gate<<<dim3(TSEQ, NH_), 64, 0, stream>>>(hf, gf, sO_l, hgob);

    // output projection with fused residual add; refresh bf16 mirror of x
    k_gemm_bf16<<<gProj, 256, 0, stream>>>(hgob, wob + wofs, x, xb, x, nullptr, TSEQ, DE_, DE_);
  }

  // 4) logits = exp(s_F) * x @ unit(embed)^T  (M=2048, N=32000, K=1024)
  const dim3 gLog(TSEQ / 128, VOCAB_ / 64);   // (16,500)
  k_gemm_bf16<<<gLog, 256, 0, stream>>>(xb, wn, (float*)d_out, nullptr, nullptr, sF, TSEQ, VOCAB_, DE_);
}
